// UNet3THM_69415261438237
// MI455X (gfx1250) — matrix operand order repaired and checked by reading, not compiled, not run on hardware
//
#include <hip/hip_runtime.h>
#include <hip/hip_bf16.h>
#include <stdint.h>

typedef __bf16 bf16;
typedef __bf16 v16bf __attribute__((ext_vector_type(16)));
typedef __bf16 v8bf  __attribute__((ext_vector_type(8)));
typedef float  v8f   __attribute__((ext_vector_type(8)));

// ---------------------------------------------------------------------------
// Activations: NDHWC, channels innermost, C padded to multiple of 32 (only x).
// Weights: pre-packed in A-fragment order (k = 16*khalf + i) [ctile][tap][chunk][lane][16],
//          staged per (ctile, kd) into LDS via GLOBAL_LOAD_ASYNC_TO_LDS_B128,
//          read back as ds_load_b128.
// MODE: 0 = conv s1, 1 = conv s2, 2 = tconv s2 (pad_a=2, parity), 3 = tconv s1
// NX:   16-wide x-tiles per wave (A-fragment reuse).  CH: K-chunks (CinP/32).
// ---------------------------------------------------------------------------
template<int MODE, int NX, int CH>
__global__ __launch_bounds__(256)
void wmma_conv_kernel(const bf16* __restrict__ in, const bf16* __restrict__ wpk,
                      const bf16* __restrict__ zbuf,
                      bf16* __restrict__ outb, float* __restrict__ outf,
                      int Cout,
                      int Din, int Hin, int Win,
                      int Dout, int Hout, int Wout,
                      int Nimg, int outCbase, int outCtot, int outOff)
{
    __shared__ bf16 smem[9 * CH * 512];            // one kd-slice of packed weights

    const int lane  = threadIdx.x & 31;
    const int khalf = lane >> 4;
    const int n     = lane & 15;
    const int wid   = threadIdx.x >> 5;            // wave in block (8 waves)

    const int wtiles  = (Wout >> 4) / NX;
    const int sptiles = Dout * Hout * wtiles;      // guaranteed divisible by 8

    int b = blockIdx.x;
    const int sblk  = b % (sptiles >> 3); b /= (sptiles >> 3);
    const int img   = b % Nimg;           b /= Nimg;
    const int ctile = b;                           // uniform across the block

    int st = sblk * 8 + wid;
    const int xt = st % wtiles; st /= wtiles;
    const int yo = st % Hout;   st /= Hout;
    const int zo = st;

    const int m0 = ctile * 16;
    const int x0 = xt * 16 * NX;
    const int CinP = CH * 32;

    v8f acc0 = {0.f,0.f,0.f,0.f,0.f,0.f,0.f,0.f};
    v8f acc1 = {0.f,0.f,0.f,0.f,0.f,0.f,0.f,0.f};

    for (int kd = 0; kd < 3; ++kd) {
        // ---- async DMA stage of this kd-slice (9 taps) into LDS, no VGPR bounce
        __syncthreads();
        {
            const bf16* src = wpk + ((size_t)(ctile * 27 + kd * 9) * CH) * 512;
            for (int i = threadIdx.x; i < 576 * CH; i += 256) {
                unsigned           ldsa = (unsigned)(uintptr_t)(smem + (size_t)i * 8);
                unsigned long long ga   = (unsigned long long)(uintptr_t)(src + (size_t)i * 8);
                asm volatile("global_load_async_to_lds_b128 %0, %1, off"
                             :: "v"(ldsa), "v"(ga) : "memory");
            }
            asm volatile("s_wait_asynccnt 0" ::: "memory");
        }
        __syncthreads();

        // ---- per-wave z coordinate (wave-uniform skip)
        int iz; bool zok = true;
        if (MODE == 0 || MODE == 3)      { iz = zo + kd - 1; }
        else if (MODE == 1)              { iz = zo * 2 + kd - 1; }
        else { int tz = zo + kd - 2; zok = (tz >= 0) && !(tz & 1); iz = tz >> 1; }
        zok = zok && (iz >= 0) && (iz < Din);
        if (!zok) continue;

        for (int kh = 0; kh < 3; ++kh) {
            int iy; bool yok = true;
            if (MODE == 0 || MODE == 3)  { iy = yo + kh - 1; }
            else if (MODE == 1)          { iy = yo * 2 + kh - 1; }
            else { int ty = yo + kh - 2; yok = (ty >= 0) && !(ty & 1); iy = ty >> 1; }
            yok = yok && (iy >= 0) && (iy < Hin);
            if (!yok) continue;

            const int rowvox = ((img * Din + iz) * Hin + iy) * Win;

            #pragma unroll
            for (int kw = 0; kw < 3; ++kw) {
                // per-lane x coords; invalid -> zero buffer (branch-free)
                int px0 = x0 + n, px1 = px0 + 16;
                int ix0, ix1; bool ok0 = true, ok1 = true;
                if (MODE == 0 || MODE == 3)  { ix0 = px0 + kw - 1;     ix1 = px1 + kw - 1; }
                else if (MODE == 1)          { ix0 = px0 * 2 + kw - 1; ix1 = px1 * 2 + kw - 1; }
                else {
                    int t0 = px0 + kw - 2, t1 = px1 + kw - 2;
                    ok0 = (t0 >= 0) && !(t0 & 1); ix0 = t0 >> 1;
                    ok1 = (t1 >= 0) && !(t1 & 1); ix1 = t1 >> 1;
                }
                ok0 = ok0 && (ix0 >= 0) && (ix0 < Win);
                ok1 = ok1 && (ix1 >= 0) && (ix1 < Win);

                const bf16* p0 = ok0 ? (in + (size_t)(rowvox + ix0) * CinP) : zbuf;
                const bf16* p1 = ok1 ? (in + (size_t)(rowvox + ix1) * CinP) : zbuf;
                const bf16* wa = smem + (size_t)((kh * 3 + kw) * CH) * 512 + lane * 16;

                #pragma unroll
                for (int c = 0; c < CH; ++c) {
                    v16bf afrag = *(const v16bf*)(wa + c * 512);        // ds_load_b128 x2
                    v16bf b0    = *(const v16bf*)(p0 + (c << 5) + (khalf << 4));
                    acc0 = __builtin_amdgcn_wmma_f32_16x16x32_bf16(
                               false, afrag, false, b0, (short)0, acc0, false, false);
                    if (NX == 2) {
                        v16bf b1 = *(const v16bf*)(p1 + (c << 5) + (khalf << 4));
                        acc1 = __builtin_amdgcn_wmma_f32_16x16x32_bf16(
                                   false, afrag, false, b1, (short)0, acc1, false, false);
                    }
                }
            }
        }
    }

    // ---- store: D VGPR r -> channel m0 + khalf*8 + r, column n -> voxel x
    #pragma unroll
    for (int j = 0; j < NX; ++j) {
        v8f a = (j == 0) ? acc0 : acc1;
        int px = x0 + j * 16 + n;
        if (outb) {   // NDHWC bf16: 8 consecutive channels = one 16B store
            int voxel = ((img * Dout + zo) * Hout + yo) * Wout + px;
            v8bf o;
            #pragma unroll
            for (int r = 0; r < 8; ++r) o[r] = (bf16)a[r];
            *(v8bf*)(outb + (size_t)voxel * outCtot + outCbase + m0 + khalf * 8) = o;
        } else {      // NCDHW f32 straight into d_out (final heads)
            int HWo = Hout * Wout;
            #pragma unroll
            for (int r = 0; r < 8; ++r) {
                int cout = m0 + khalf * 8 + r;
                if (cout >= Cout) continue;
                int oidx = ((img * outCtot + cout) * Dout + zo) * HWo + yo * Wout + px;
                outf[outOff + oidx] = a[r];
            }
        }
    }
}

// ---------------------------------------------------------------------------
// Pack f32 weights (OIDHW or DHWIO) into bf16 A-fragment layout, zero-padded.
__global__ void pack_weights_kernel(const float* __restrict__ w, bf16* __restrict__ pk,
                                    int Cin, int Cout, int tconv, int chunks, int ctiles)
{
    int idx = blockIdx.x * blockDim.x + threadIdx.x;
    int total = ctiles * 27 * chunks * 512;
    if (idx >= total) return;
    int i     = idx & 15;
    int lane  = (idx >> 4) & 31;
    int rest  = idx >> 9;
    int chunk = rest % chunks; rest /= chunks;
    int tap   = rest % 27;     rest /= 27;
    int ctile = rest;
    int khalf = lane >> 4, m = lane & 15;
    int cout  = ctile * 16 + m;
    int kk    = khalf * 16 + i;
    int ci    = chunk * 32 + kk;
    float v = 0.f;
    if (cout < Cout && ci < Cin)
        v = tconv ? w[(tap * Cin + ci) * Cout + cout]
                  : w[(cout * Cin + ci) * 27 + tap];
    pk[idx] = (bf16)v;
}

// NCDHW f32 x -> NDHWC bf16 padded to CP channels (zeros above Cin)
__global__ void cvt_x_kernel(const float* __restrict__ x, bf16* __restrict__ out,
                             int Cin, int CP, int DHW, int Nimg)
{
    long long idx = (long long)blockIdx.x * blockDim.x + threadIdx.x;
    long long total = (long long)Nimg * DHW * CP;
    if (idx >= total) return;
    int c   = (int)(idx % CP);
    long long v = idx / CP;
    int sp  = (int)(v % DHW);
    int img = (int)(v / DHW);
    out[idx] = (c < Cin) ? (bf16)x[((long long)(img * Cin + c)) * DHW + sp] : (bf16)0.f;
}

__global__ void zero_f32_kernel(float* p, int n)
{
    int i = blockIdx.x * blockDim.x + threadIdx.x;
    if (i < n) p[i] = 0.f;
}

// per-channel sum/sumsq over NDHWC buffer (channel-coalesced)
__global__ __launch_bounds__(256)
void bn_stats_kernel(const bf16* __restrict__ buf, float* __restrict__ sum,
                     float* __restrict__ sq,
                     int C, int cbase, int Ctot, int DHW, int Nimg)
{
    __shared__ float ls[128], ls2[128];
    if ((int)threadIdx.x < C) { ls[threadIdx.x] = 0.f; ls2[threadIdx.x] = 0.f; }
    __syncthreads();

    const int groups = blockDim.x / C;
    const int c      = threadIdx.x % C;
    const int grp    = threadIdx.x / C;
    const long long totalVox = (long long)Nimg * DHW;

    float s = 0.f, s2 = 0.f;
    if (grp < groups) {
        for (long long vox = (long long)blockIdx.x * groups + grp; vox < totalVox;
             vox += (long long)gridDim.x * groups) {
            float v = (float)buf[vox * Ctot + cbase + c];
            s += v; s2 += v * v;
        }
        atomicAdd(&ls[c], s);
        atomicAdd(&ls2[c], s2);
    }
    __syncthreads();
    if ((int)threadIdx.x < C) {
        atomicAdd(&sum[threadIdx.x], ls[threadIdx.x]);
        atomicAdd(&sq[threadIdx.x],  ls2[threadIdx.x]);
    }
}

__global__ __launch_bounds__(256)
void bn_apply_kernel(bf16* __restrict__ buf, const float* __restrict__ sum,
                     const float* __restrict__ sq, const float* __restrict__ g,
                     const float* __restrict__ b,
                     int C, int cbase, int Ctot, int DHW, int Nimg)
{
    long long idx = (long long)blockIdx.x * blockDim.x + threadIdx.x;
    long long total = (long long)Nimg * DHW * C;
    if (idx >= total) return;
    int c = (int)(idx % C);
    long long vox = idx / C;
    float cnt = (float)Nimg * (float)DHW;
    float m   = sum[c] / cnt;
    float var = sq[c] / cnt - m * m;
    long long o = vox * Ctot + cbase + c;
    float v = ((float)buf[o] - m) * rsqrtf(var + 1e-5f) * g[c] + b[c];
    buf[o] = (bf16)fmaxf(v, 0.f);
}

// channel-concat copy in NDHWC
__global__ void copy_channels_kernel(const bf16* __restrict__ src, bf16* __restrict__ dst,
                                     int Cs, int Ctot, int cbase, int DHW, int Nimg)
{
    long long idx = (long long)blockIdx.x * blockDim.x + threadIdx.x;
    long long total = (long long)Nimg * DHW * Cs;
    if (idx >= total) return;
    int c = (int)(idx % Cs);
    long long vox = idx / Cs;
    dst[vox * Ctot + cbase + c] = src[vox * Cs + c];
}

// ---------------------------------------------------------------------------
template<int MODE, int NX, int CH>
static inline void conv_launch_t(hipStream_t s, const bf16* in, const bf16* wpk,
                                 const bf16* zbuf, bf16* outb, float* outf, int Cout,
                                 int Din, int Hin, int Win, int Dout, int Hout, int Wout,
                                 int Nimg, int cbase, int ctot, int outOff)
{
    int sptiles = Dout * Hout * ((Wout / 16) / NX);
    int blocks  = ((Cout + 15) / 16) * Nimg * (sptiles / 8);
    wmma_conv_kernel<MODE, NX, CH><<<blocks, 256, 0, s>>>(in, wpk, zbuf, outb, outf, Cout,
                                                          Din, Hin, Win, Dout, Hout, Wout,
                                                          Nimg, cbase, ctot, outOff);
}

static inline void pack_launch(hipStream_t s, const float* w, bf16* pk,
                               int Cin, int Cout, int tconv)
{
    int chunks = (Cin + 31) / 32, ctiles = (Cout + 15) / 16;
    int total = ctiles * 27 * chunks * 512;
    pack_weights_kernel<<<(total + 255) / 256, 256, 0, s>>>(w, pk, Cin, Cout, tconv, chunks, ctiles);
}

static inline void bn_launch(hipStream_t s, bf16* buf, float* sum, float* sq,
                             const float* g, const float* b,
                             int C, int cbase, int ctot, int DHW, int Nimg)
{
    zero_f32_kernel<<<1, 256, 0, s>>>(sum, C);
    zero_f32_kernel<<<1, 256, 0, s>>>(sq,  C);
    bn_stats_kernel<<<1024, 256, 0, s>>>(buf, sum, sq, C, cbase, ctot, DHW, Nimg);
    long long tot = (long long)Nimg * DHW * C;
    bn_apply_kernel<<<(int)((tot + 255) / 256), 256, 0, s>>>(buf, sum, sq, g, b,
                                                             C, cbase, ctot, DHW, Nimg);
}

// ---------------------------------------------------------------------------
extern "C" void kernel_launch(void* const* d_in, const int* in_sizes, int n_in,
                              void* d_out, int out_size, void* d_ws, size_t ws_size,
                              hipStream_t stream)
{
    const float* x    = (const float*)d_in[0];
    const float* w0   = (const float*)d_in[1];
    const float* g0   = (const float*)d_in[2];
    const float* b0   = (const float*)d_in[3];
    const float* w1   = (const float*)d_in[4];
    const float* g1   = (const float*)d_in[5];
    const float* b1   = (const float*)d_in[6];
    const float* w2   = (const float*)d_in[7];
    const float* g2   = (const float*)d_in[8];
    const float* b2   = (const float*)d_in[9];
    const float* wt2r = (const float*)d_in[10];
    const float* g2r  = (const float*)d_in[11];
    const float* b2r  = (const float*)d_in[12];
    const float* wt1r = (const float*)d_in[13];
    const float* g1r  = (const float*)d_in[14];
    const float* b1r  = (const float*)d_in[15];
    const float* wt0r = (const float*)d_in[16];
    const float* wt2c = (const float*)d_in[17];
    const float* g2c  = (const float*)d_in[18];
    const float* b2c  = (const float*)d_in[19];
    const float* wt1c = (const float*)d_in[20];
    const float* g1c  = (const float*)d_in[21];
    const float* b1c  = (const float*)d_in[22];
    const float* wt0c = (const float*)d_in[23];

    float* out = (float*)d_out;
    char*  ws  = (char*)d_ws;
    size_t off = 0;
    auto alloc = [&](size_t bytes) { size_t o = off; off += (bytes + 255) & ~(size_t)255; return o; };

    float* stSum = (float*)(ws + alloc(128 * sizeof(float)));
    float* stSq  = (float*)(ws + alloc(128 * sizeof(float)));
    bf16*  zbuf  = (bf16*)(ws + alloc(256));          // 256B of zeros (max CinP row)

    // packed bf16 weights: ctiles * 27 * chunks * 512 elements
    bf16* w0p   = (bf16*)(ws + alloc((size_t)27648  * 2));
    bf16* w1p   = (bf16*)(ws + alloc((size_t)55296  * 2));
    bf16* w2p   = (bf16*)(ws + alloc((size_t)221184 * 2));
    bf16* wt2rp = (bf16*)(ws + alloc((size_t)221184 * 2));
    bf16* wt1rp = (bf16*)(ws + alloc((size_t)55296  * 2));
    bf16* wt0rp = (bf16*)(ws + alloc((size_t)13824  * 2));
    bf16* wt2cp = (bf16*)(ws + alloc((size_t)221184 * 2));
    bf16* wt1cp = (bf16*)(ws + alloc((size_t)221184 * 2));
    bf16* wt0cp = (bf16*)(ws + alloc((size_t)41472  * 2));

    // NDHWC bf16 activations
    const int D64 = 262144, D32 = 32768, D16 = 4096;
    bf16* xb     = (bf16*)(ws + alloc((size_t)2 * D64 * 32  * 2));  // padded 16->32
    bf16* s1b    = (bf16*)(ws + alloc((size_t)2 * D64 * 32  * 2));
    bf16* s2b    = (bf16*)(ws + alloc((size_t)2 * D32 * 64  * 2));
    bf16* s4b    = (bf16*)(ws + alloc((size_t)2 * D16 * 128 * 2));
    bf16* rec32b = (bf16*)(ws + alloc((size_t)2 * D32 * 64  * 2));
    bf16* rec64b = (bf16*)(ws + alloc((size_t)2 * D64 * 32  * 2));
    bf16* cl32   = (bf16*)(ws + alloc((size_t)2 * D32 * 128 * 2));
    bf16* cl64   = (bf16*)(ws + alloc((size_t)2 * D64 * 96  * 2));

    // --- preparation
    zero_f32_kernel<<<1, 256, 0, stream>>>((float*)zbuf, 64);
    {
        long long tot = (long long)2 * D64 * 32;
        cvt_x_kernel<<<(int)((tot + 255) / 256), 256, 0, stream>>>(x, xb, 16, 32, D64, 2);
    }
    pack_launch(stream, w0,   w0p,   16,  32, 0);
    pack_launch(stream, w1,   w1p,   32,  64, 0);
    pack_launch(stream, w2,   w2p,   64, 128, 0);
    pack_launch(stream, wt2r, wt2rp, 128, 64, 1);
    pack_launch(stream, wt1r, wt1rp, 64,  32, 1);
    pack_launch(stream, wt0r, wt0rp, 32,  16, 1);
    pack_launch(stream, wt2c, wt2cp, 128, 64, 1);
    pack_launch(stream, wt1c, wt1cp, 128, 64, 1);
    pack_launch(stream, wt0c, wt0cp, 96,   2, 1);

    // --- encoder
    conv_launch_t<0,2,1>(stream, xb,  w0p, zbuf, s1b, nullptr, 32,  64,64,64, 64,64,64, 2, 0, 32, 0);
    bn_launch(stream, s1b, stSum, stSq, g0, b0, 32, 0, 32, D64, 2);

    conv_launch_t<1,2,1>(stream, s1b, w1p, zbuf, s2b, nullptr, 64,  64,64,64, 32,32,32, 2, 0, 64, 0);
    bn_launch(stream, s2b, stSum, stSq, g1, b1, 64, 0, 64, D32, 2);

    conv_launch_t<1,1,2>(stream, s2b, w2p, zbuf, s4b, nullptr, 128, 32,32,32, 16,16,16, 2, 0, 128, 0);
    bn_launch(stream, s4b, stSum, stSq, g2, b2, 128, 0, 128, D16, 2);

    // --- level 2 decoders (16 -> 32)
    conv_launch_t<2,2,4>(stream, s4b, wt2rp, zbuf, rec32b, nullptr, 64, 16,16,16, 32,32,32, 2, 0, 64, 0);
    bn_launch(stream, rec32b, stSum, stSq, g2r, b2r, 64, 0, 64, D32, 2);

    conv_launch_t<2,2,4>(stream, s4b, wt2cp, zbuf, cl32, nullptr, 64, 16,16,16, 32,32,32, 2, 0, 128, 0);
    bn_launch(stream, cl32, stSum, stSq, g2c, b2c, 64, 0, 128, D32, 2);
    {   // concat [cl, s2]
        long long tot = (long long)2 * D32 * 64;
        copy_channels_kernel<<<(int)((tot + 255) / 256), 256, 0, stream>>>(s2b, cl32, 64, 128, 64, D32, 2);
    }

    // --- level 1 decoders (32 -> 64)
    conv_launch_t<2,2,2>(stream, rec32b, wt1rp, zbuf, rec64b, nullptr, 32, 32,32,32, 64,64,64, 2, 0, 32, 0);
    bn_launch(stream, rec64b, stSum, stSq, g1r, b1r, 32, 0, 32, D64, 2);

    conv_launch_t<2,2,4>(stream, cl32, wt1cp, zbuf, cl64, nullptr, 64, 32,32,32, 64,64,64, 2, 0, 96, 0);
    bn_launch(stream, cl64, stSum, stSq, g1c, b1c, 64, 0, 96, D64, 2);
    {   // concat [cl, s1]
        long long tot = (long long)2 * D64 * 32;
        copy_channels_kernel<<<(int)((tot + 255) / 256), 256, 0, stream>>>(s1b, cl64, 32, 96, 64, D64, 2);
    }

    // --- final stride-1 tconvs, f32 straight to d_out: cl first, then rec
    const int recOff = 2 * 2 * D64;
    conv_launch_t<3,2,3>(stream, cl64,   wt0cp, zbuf, nullptr, out, 2,  64,64,64, 64,64,64, 2, 0, 2,  0);
    conv_launch_t<3,2,1>(stream, rec64b, wt0rp, zbuf, nullptr, out, 16, 64,64,64, 64,64,64, 2, 0, 16, recOff);
}
